// DualContextAttention_57552561767193
// MI455X (gfx1250) — matrix operand order repaired and checked by reading, not compiled, not run on hardware
//
#include <hip/hip_runtime.h>
#include <hip/hip_bf16.h>

// ---------------------------------------------------------------------------
// DualContextAttention forward for MI455X (gfx1250, wave32, WMMA + TDM).
// All heavy math runs through v_wmma_f32_16x16x32_f16 (f16 in, f32 accum).
// Attention is flash-style fused (no NxN energy materialization); K/V tiles
// are staged into LDS by the Tensor Data Mover (tensor_load_to_lds) with
// hardware row padding providing the bank-conflict-free LDS strides.
// ---------------------------------------------------------------------------

typedef __attribute__((ext_vector_type(16))) _Float16 v16h;
typedef __attribute__((ext_vector_type(8)))  _Float16 v8h;
typedef __attribute__((ext_vector_type(8)))  float    v8f;

typedef __attribute__((ext_vector_type(4))) unsigned int u32x4;
typedef __attribute__((ext_vector_type(4))) int          i32x4;
typedef __attribute__((ext_vector_type(8))) int          i32x8;

#define LDS_FENCE() asm volatile("s_wait_dscnt 0x0" ::: "memory")

#if defined(__has_builtin)
#if __has_builtin(__builtin_amdgcn_tensor_load_to_lds)
#define HAVE_TDM 1
#endif
#endif

// ---- WMMA fragment helpers (layouts per CDNA5 ISA 7.12.2, wave32) ----------
// A (16x32 f16): lane l -> row M = l&15; halves 0..7 = K (8*g ..), 8..15 = K (16+8*g ..)
__device__ __forceinline__ v16h frag_a(const _Float16* p, int g) {
  v8h lo = *(const v8h*)(p + 8 * g);
  v8h hi = *(const v8h*)(p + 16 + 8 * g);
  v16h r;
#pragma unroll
  for (int i = 0; i < 8; ++i) { r[i] = lo[i]; r[i + 8] = hi[i]; }
  return r;
}
// B (32x16 f16): lane l -> col N = l&15; halves 0..7 = K (8*g ..), 8..15 = K (16+8*g ..),
// assuming memory layout [N][K] (row = output column, contiguous in K).
__device__ __forceinline__ v16h frag_b(const _Float16* p, int g) {
  v8h lo = *(const v8h*)(p + 8 * g);
  v8h hi = *(const v8h*)(p + 16 + 8 * g);
  v16h r;
#pragma unroll
  for (int i = 0; i < 8; ++i) { r[i] = lo[i]; r[i + 8] = hi[i]; }
  return r;
}
__device__ __forceinline__ v16h zero16h() {
  v16h z;
#pragma unroll
  for (int i = 0; i < 16; ++i) z[i] = (_Float16)0;
  return z;
}
__device__ __forceinline__ v8f zero8f() {
  v8f z;
#pragma unroll
  for (int i = 0; i < 8; ++i) z[i] = 0.f;
  return z;
}
__device__ __forceinline__ v8f wmma16(v16h a, v16h b, v8f c) {
  return __builtin_amdgcn_wmma_f32_16x16x32_f16(false, a, false, b, (short)0, c,
                                                false, false);
}

// ---- TDM: 2-D tile of 2-byte elements, global -> LDS, with LDS row padding.
// pad_interval_code: pad after 2^(code+1) DWORDs; pad_amount_code: (code+1) DWORDs.
__device__ __forceinline__ void tdm_load_2d(unsigned lds_addr, const void* gaddr,
                                            unsigned tile_d0, unsigned tile_d1,
                                            unsigned stride_d0,
                                            unsigned pad_interval_code,
                                            unsigned pad_amount_code) {
#ifdef HAVE_TDM
  unsigned long long ga = (unsigned long long)(uintptr_t)gaddr;
  u32x4 g0;
  g0[0] = 1u;                                   // count=1, user mode, no gather
  g0[1] = lds_addr;                             // D# lds_addr (bytes)
  g0[2] = (unsigned)(ga & 0xFFFFFFFFu);         // global_addr[31:0]
  g0[3] = (unsigned)((ga >> 32) & 0x01FFFFFFu)  // global_addr[56:32]
          | (2u << 30);                         // type = 2 ("image")
  i32x8 g1;
  g1[0] = (int)((1u << 16)                      // data_size = 2 bytes
                | (1u << 20)                    // pad_enable
                | (pad_interval_code << 22) | (pad_amount_code << 25));
  g1[1] = (int)((tile_d0 & 0xFFFFu) << 16);     // tensor_dim0[15:0] (== tile)
  g1[2] = (int)(((tile_d0 >> 16) & 0xFFFFu) | ((tile_d1 & 0xFFFFu) << 16));
  g1[3] = (int)(((tile_d1 >> 16) & 0xFFFFu) | ((tile_d0 & 0xFFFFu) << 16));
  g1[4] = (int)(tile_d1 & 0xFFFFu);             // tile_dim1 (tile_dim2 = 0)
  g1[5] = (int)stride_d0;                       // tensor_dim0_stride[31:0]
  g1[6] = 0;
  g1[7] = 0;
  i32x4 z4 = {0, 0, 0, 0};
#if defined(__clang_major__) && __clang_major__ >= 23
  i32x8 z8 = {0, 0, 0, 0, 0, 0, 0, 0};
  __builtin_amdgcn_tensor_load_to_lds(g0, g1, z4, z4, z8, 0);
#else
  __builtin_amdgcn_tensor_load_to_lds(g0, g1, z4, z4, 0);
#endif
#else
  (void)lds_addr; (void)gaddr; (void)tile_d0; (void)tile_d1;
  (void)stride_d0; (void)pad_interval_code; (void)pad_amount_code;
#endif
}

// ---------------------------------------------------------------------------
// Kernel 0: weight prep (f32->f16 conversions, BN folding, 3x3 tap re-layout)
// ---------------------------------------------------------------------------
struct PrepArgs {
  const float *wq, *wk, *wv, *bq, *bk, *bv;
  const float *d1w, *l1w, *l2w, *l3w;
  const float *bng[4], *bnb[4], *bnm[4], *bnv[4];   // bn1, lbn1, lbn2, lbn3
  const float *rbng, *rbnb, *rbnm, *rbnv;
  _Float16 *wqkv, *d1h, *l1h, *l2tap, *l3h;
  float *biasqkv, *bns, *bnt;                        // bns/bnt: [5][256]
};

__global__ void k_prep(PrepArgs a) {
  int i0 = blockIdx.x * blockDim.x + threadIdx.x;
  int stride = gridDim.x * blockDim.x;
  for (int i = i0; i < 512 * 256; i += stride) {
    int oc = i >> 8, c = i & 255;
    float w = oc < 128   ? a.wq[oc * 256 + c]
              : oc < 256 ? a.wk[(oc - 128) * 256 + c]
                         : a.wv[(oc - 256) * 256 + c];
    a.wqkv[i] = (_Float16)w;
  }
  for (int i = i0; i < 512; i += stride)
    a.biasqkv[i] = i < 128 ? a.bq[i] : i < 256 ? a.bk[i - 128] : a.bv[i - 256];
  for (int i = i0; i < 65536; i += stride) {
    a.d1h[i] = (_Float16)a.d1w[i];
    a.l1h[i] = (_Float16)a.l1w[i];
    a.l3h[i] = (_Float16)a.l3w[i];
  }
  for (int i = i0; i < 9 * 65536; i += stride) {
    int tap = i >> 16, r = i & 65535;
    a.l2tap[i] = (_Float16)a.l2w[(size_t)r * 9 + tap];  // [O][C][3][3] -> [tap][O][C]
  }
  for (int i = i0; i < 256; i += stride) {
#pragma unroll
    for (int k = 0; k < 4; ++k) {
      float s = a.bng[k][i] * rsqrtf(a.bnv[k][i] + 1e-5f);
      a.bns[k * 256 + i] = s;
      a.bnt[k * 256 + i] = a.bnb[k][i] - a.bnm[k][i] * s;
    }
    if (i < 128) {
      float s = a.rbng[i] * rsqrtf(a.rbnv[i] + 1e-5f);
      a.bns[4 * 256 + i] = s;
      a.bnt[4 * 256 + i] = a.rbnb[i] - a.rbnm[i] * s;
    }
  }
}

// ---------------------------------------------------------------------------
// Kernel 1: x [B][C][N] f32  ->  xT [B*N][C] f16  (GEMM B-operand layout)
// ---------------------------------------------------------------------------
__global__ void k_xpose(const float* __restrict__ x, _Float16* __restrict__ xT) {
  size_t i = (size_t)blockIdx.x * blockDim.x + threadIdx.x;
  size_t stride = (size_t)gridDim.x * blockDim.x;
  for (; i < (size_t)16384 * 256; i += stride) {
    int c = (int)(i & 255);
    size_t bn = i >> 8;
    int b = (int)(bn >> 12);
    int pix = (int)(bn & 4095);
    xT[i] = (_Float16)x[(((size_t)(b * 256 + c)) << 12) + pix];
  }
}

// ---------------------------------------------------------------------------
// Kernel 2: generic 1x1-conv GEMM.  One wave computes a 16(oc) x 16(pixel)
// tile, K-loop in steps of 32 through v_wmma_f32_16x16x32_f16.
// ---------------------------------------------------------------------------
enum { M_QKV = 0, M_GATE = 1, M_BNRELU_H = 2, M_BNRELU_F = 3 };

__global__ __launch_bounds__(32) void k_gemm(
    const _Float16* __restrict__ Wh,   // [OC][K] f16
    const _Float16* __restrict__ Xh,   // [B*N][K] f16
    const float* __restrict__ bias,    // [OC]
    const float* __restrict__ bns, const float* __restrict__ bnt,
    _Float16* __restrict__ outH, float* __restrict__ outF,
    const _Float16* __restrict__ auxH,                       // feat (GATE mode)
    _Float16* __restrict__ qOut, _Float16* __restrict__ kOut,
    _Float16* __restrict__ vOut,
    int K, int mode) {
  const int lane = threadIdx.x & 31;
  const int g = lane >> 4, l15 = lane & 15;
  const int n0 = blockIdx.x * 16;
  const int oc0 = blockIdx.y * 16;
  const int OC = gridDim.y * 16;

  v8f acc = zero8f();
  const _Float16* wrow = Wh + (size_t)(oc0 + l15) * K;
  const _Float16* xrow = Xh + (size_t)(n0 + l15) * K;
  for (int kk = 0; kk < K; kk += 32) {
    v16h a = frag_a(wrow + kk, g);
    v16h b = frag_b(xrow + kk, g);
    acc = wmma16(a, b, acc);
  }

  const int n = n0 + l15;          // D layout: lane -> col n, vgpr j -> row j+8g
#pragma unroll
  for (int j = 0; j < 8; ++j) {
    int oc = oc0 + j + 8 * g;
    float v = acc[j] + bias[oc];
    if (mode == M_QKV) {
      if (oc < 128)
        qOut[(size_t)n * 128 + oc] = (_Float16)v;
      else if (oc < 256)
        kOut[(size_t)n * 128 + (oc - 128)] = (_Float16)v;
      else {
        int b = n >> 12, pix = n & 4095;
        vOut[(((size_t)(b * 256 + (oc - 256))) << 12) + pix] = (_Float16)v;
      }
    } else if (mode == M_GATE) {
      float t = fmaxf(v * bns[oc] + bnt[oc], 0.f);
      float gate = 1.f / (1.f + __expf(-t));
      float f = (float)auxH[(size_t)n * 256 + oc];
      outH[(size_t)n * 256 + oc] = (_Float16)(f * gate);     // y = feat*gate1
    } else if (mode == M_BNRELU_H) {
      float t = fmaxf(v * bns[oc] + bnt[oc], 0.f);
      outH[(size_t)n * OC + oc] = (_Float16)t;
    } else {  // M_BNRELU_F: write f32 [B][C][N]
      float t = fmaxf(v * bns[oc] + bnt[oc], 0.f);
      int b = n >> 12, pix = n & 4095;
      outF[(((size_t)(b * 256 + oc)) << 12) + pix] = t;
    }
  }
}

// ---------------------------------------------------------------------------
// Kernel 3: flash attention.  Block = 4 waves x 16 queries; chunk = 32 keys.
// K/V chunks staged in LDS by the Tensor Data Mover (wave 0 issues, waits on
// TENSORcnt); per chunk: 8 WMMA scores, online softmax (wave32 shfl_xor in
// 16-lane groups), 16 WMMA for the 256-channel V * P^T accumulation.
// ---------------------------------------------------------------------------
#define KBS 136   // 272 B rows (128 halves + TDM pad of 4 DW): conflict-free
#define VBS 40    // 80 B rows  (32 halves + TDM pad of 4 DW)
#define PBS 40

__global__ __launch_bounds__(128) void k_attn(
    const _Float16* __restrict__ Qh,   // [B*N][128]
    const _Float16* __restrict__ Kh,   // [B*N][128]
    const _Float16* __restrict__ Vh,   // [B][256][N]
    float* __restrict__ featF,         // [B][256][N] f32
    _Float16* __restrict__ featH) {    // [B*N][256] f16
  __shared__ _Float16 kb[32 * KBS];
  __shared__ _Float16 vb[256 * VBS];
  __shared__ _Float16 pb[4 * 16 * PBS];
  __shared__ float scl[4][16];

  const int tid = threadIdx.x;
  const int wave = tid >> 5, lane = tid & 31;
  const int g = lane >> 4, l15 = lane & 15;
  const int b = blockIdx.y;
  const int q0 = blockIdx.x * 64 + wave * 16;

  // LDS byte offsets for TDM descriptors (flat LDS addr[31:0] == LDS offset)
  const unsigned kb_lds = (unsigned)(uintptr_t)(void*)kb;
  const unsigned vb_lds = (unsigned)(uintptr_t)(void*)vb;

  // preload this wave's Q fragments (16 queries x 128 channels)
  const _Float16* qrow = Qh + ((size_t)(b * 4096) + q0 + l15) * 128;
  v16h aQ[4];
#pragma unroll
  for (int kk = 0; kk < 4; ++kk) aQ[kk] = frag_a(qrow + kk * 32, g);

  v8f facc[16];
#pragma unroll
  for (int t = 0; t < 16; ++t) facc[t] = zero8f();
  float runM[8], runL[8];
#pragma unroll
  for (int j = 0; j < 8; ++j) { runM[j] = -1e30f; runL[j] = 0.f; }

  _Float16* pbw = pb + wave * 16 * PBS;

  for (int m0 = 0; m0 < 4096; m0 += 32) {
    __syncthreads();   // everyone done reading previous K/V chunk
#ifdef HAVE_TDM
    if (wave == 0) {
      // K chunk: 32 key rows x 128 ch; pad 4 DW after every 64 DW -> 136-half rows
      tdm_load_2d(kb_lds, Kh + ((size_t)(b * 4096) + m0) * 128,
                  /*tile_d0=*/128, /*tile_d1=*/32, /*stride=*/128,
                  /*pad_int=*/5, /*pad_amt=*/3);
      // V chunk: 256 ch rows x 32 keys; pad 4 DW after every 16 DW -> 40-half rows
      tdm_load_2d(vb_lds, Vh + (((size_t)(b * 256)) << 12) + m0,
                  /*tile_d0=*/32, /*tile_d1=*/256, /*stride=*/4096,
                  /*pad_int=*/3, /*pad_amt=*/3);
      __builtin_amdgcn_s_wait_tensorcnt(0);
    }
#else
    {   // fallback: manual staging
      int r = tid >> 2, col = (tid & 3) * 32;
      const v8h* src = (const v8h*)(Kh + ((size_t)(b * 4096) + m0 + r) * 128 + col);
      v8h* dst = (v8h*)(kb + r * KBS + col);
#pragma unroll
      for (int i = 0; i < 4; ++i) dst[i] = src[i];
#pragma unroll
      for (int rr = 0; rr < 2; ++rr) {
        int rv = tid * 2 + rr;
        const v8h* sv = (const v8h*)(Vh + (((size_t)(b * 256) + rv) << 12) + m0);
        v8h* dv = (v8h*)(vb + rv * VBS);
#pragma unroll
        for (int i = 0; i < 4; ++i) dv[i] = sv[i];
      }
    }
#endif
    if (m0 + 32 < 4096)
      __builtin_prefetch(Kh + ((size_t)(b * 4096) + m0 + 32) * 128, 0, 1);
    __syncthreads();   // K/V chunk visible to all waves

    // scores: two 16q x 16key tiles over K=128
    v8f s0 = zero8f(), s1 = zero8f();
#pragma unroll
    for (int kk = 0; kk < 4; ++kk) {
      v16h b0 = frag_b(kb + l15 * KBS + kk * 32, g);
      v16h b1 = frag_b(kb + (16 + l15) * KBS + kk * 32, g);
      s0 = wmma16(aQ[kk], b0, s0);
      s1 = wmma16(aQ[kk], b1, s1);
    }

    // online softmax: vgpr j of tile holds query (j + 8*g), keys across 16 lanes
    float corr[8];
#pragma unroll
    for (int j = 0; j < 8; ++j) {
      float mx = fmaxf(s0[j], s1[j]);
#pragma unroll
      for (int off = 1; off < 16; off <<= 1)
        mx = fmaxf(mx, __shfl_xor(mx, off, 32));
      float nm = fmaxf(runM[j], mx);
      float cr = __expf(runM[j] - nm);
      float p0 = __expf(s0[j] - nm);
      float p1 = __expf(s1[j] - nm);
      int q = j + 8 * g;
      pbw[q * PBS + l15] = (_Float16)p0;        // P^T staged as [query][32 keys]
      pbw[q * PBS + 16 + l15] = (_Float16)p1;
      float ps = p0 + p1;
#pragma unroll
      for (int off = 1; off < 16; off <<= 1) ps += __shfl_xor(ps, off, 32);
      runL[j] = runL[j] * cr + ps;
      runM[j] = nm;
      corr[j] = cr;
    }
    // broadcast per-query rescale to lanes (feat-acc lanes are query-indexed)
    if (l15 == 0) {
#pragma unroll
      for (int j = 0; j < 8; ++j) scl[wave][j + 8 * g] = corr[j];
    }
    LDS_FENCE();   // same-wave LDS RAW (in-order DS pipe + explicit wait)
    float myscale = scl[wave][l15];
#pragma unroll
    for (int t = 0; t < 16; ++t)
#pragma unroll
      for (int j = 0; j < 8; ++j) facc[t][j] *= myscale;

    // feat += V(16c x 32keys) * P^T(32keys x 16q), 16 channel tiles
    v16h bP = frag_b(pbw + l15 * PBS, g);
#pragma unroll
    for (int t = 0; t < 16; ++t) {
      v16h aV = frag_a(vb + (t * 16 + l15) * VBS, g);
      facc[t] = wmma16(aV, bP, facc[t]);
    }
  }

  // finalize: divide by softmax denominator and store feat (f32 CN + f16 NC)
  if (l15 == 0) {
#pragma unroll
    for (int j = 0; j < 8; ++j) scl[wave][j + 8 * g] = 1.f / runL[j];
  }
  LDS_FENCE();
  float inv = scl[wave][l15];
  int n = q0 + l15;
#pragma unroll
  for (int t = 0; t < 16; ++t) {
#pragma unroll
    for (int j = 0; j < 8; ++j) {
      int c = t * 16 + j + 8 * g;
      float fv = facc[t][j] * inv;
      featF[(((size_t)(b * 256) + c) << 12) + n] = fv;
      featH[((size_t)(b * 4096) + n) * 256 + c] = (_Float16)fv;
    }
  }
}

// ---------------------------------------------------------------------------
// Kernel 4: 3x3 conv as 9 shifted GEMM accumulations (tap-major weights).
// ---------------------------------------------------------------------------
__global__ __launch_bounds__(32) void k_conv3(
    const _Float16* __restrict__ Wtap,  // [9][256][256]
    const _Float16* __restrict__ Xh,    // [B*N][256]
    const float* __restrict__ bias, const float* __restrict__ bns,
    const float* __restrict__ bnt, _Float16* __restrict__ outH) {
  const int lane = threadIdx.x & 31;
  const int g = lane >> 4, l15 = lane & 15;
  const int n0 = blockIdx.x * 16, oc0 = blockIdx.y * 16;
  const int n = n0 + l15;
  const int pix = n & 4095, px = pix & 63, py = (pix >> 6) & 63;

  v8f acc = zero8f();
#pragma unroll
  for (int tap = 0; tap < 9; ++tap) {
    int dy = tap / 3 - 1, dx = tap % 3 - 1;
    bool valid = ((unsigned)(px + dx) < 64u) && ((unsigned)(py + dy) < 64u);
    int nsrc = valid ? (n + dy * 64 + dx) : n;   // clamp to keep address legal
    const _Float16* wrow = Wtap + (size_t)tap * 65536 + (size_t)(oc0 + l15) * 256;
    const _Float16* xrow = Xh + (size_t)nsrc * 256;
    for (int kk = 0; kk < 256; kk += 32) {
      v16h a = frag_a(wrow + kk, g);
      v16h b = frag_b(xrow + kk, g);
      if (!valid) b = zero16h();                 // per-lane zero pad (EXEC stays full)
      acc = wmma16(a, b, acc);
    }
  }
#pragma unroll
  for (int j = 0; j < 8; ++j) {
    int oc = oc0 + j + 8 * g;
    float t = fmaxf((acc[j] + bias[oc]) * bns[oc] + bnt[oc], 0.f);
    outH[(size_t)n * 256 + oc] = (_Float16)t;
  }
}

// ---------------------------------------------------------------------------
// Kernel 5/6: SE branch (global-avg-pool + two tiny FCs)
// ---------------------------------------------------------------------------
__global__ void k_pool(const _Float16* __restrict__ yNC, float* __restrict__ pooled) {
  int b = blockIdx.x, c = threadIdx.x;   // 256 threads, coalesced across c
  const _Float16* p = yNC + ((size_t)b << 12) * 256 + c;
  float s = 0.f;
  for (int n = 0; n < 4096; ++n) s += (float)p[(size_t)n * 256];
  pooled[b * 256 + c] = s * (1.f / 4096.f);
}

__global__ void k_se(const float* __restrict__ pooled,
                     const float* __restrict__ r1w, const float* __restrict__ r1b,
                     const float* __restrict__ bns, const float* __restrict__ bnt,
                     const float* __restrict__ r2w, const float* __restrict__ r2b,
                     float* __restrict__ yr) {
  __shared__ float p[256];
  __shared__ float h1[128];
  int b = blockIdx.x, t = threadIdx.x;
  p[t] = pooled[b * 256 + t];
  __syncthreads();
  if (t < 128) {
    float s = 0.f;
    for (int c = 0; c < 256; ++c) s += r1w[t * 256 + c] * p[c];
    s += r1b[t];
    s = s * bns[4 * 256 + t] + bnt[4 * 256 + t];
    h1[t] = fmaxf(s, 0.f);
  }
  __syncthreads();
  float s = 0.f;
  for (int c = 0; c < 128; ++c) s += r2w[t * 128 + c] * h1[c];
  yr[b * 256 + t] = s + r2b[t];
}

// ---------------------------------------------------------------------------
// Kernel 7: out = feat * sigmoid(yl + yr)
// ---------------------------------------------------------------------------
__global__ void k_final(const float* __restrict__ featF,
                        const float* __restrict__ ylF,
                        const float* __restrict__ yr, float* __restrict__ out) {
  size_t i = (size_t)blockIdx.x * blockDim.x + threadIdx.x;
  size_t stride = (size_t)gridDim.x * blockDim.x;
  for (; i < (size_t)4 * 256 * 4096; i += stride) {
    int bc = (int)(i >> 12);
    float s = ylF[i] + yr[bc];
    out[i] = featF[i] * (1.f / (1.f + __expf(-s)));
  }
}

// ---------------------------------------------------------------------------
// Host launcher
// ---------------------------------------------------------------------------
extern "C" void kernel_launch(void* const* d_in, const int* in_sizes, int n_in,
                              void* d_out, int out_size, void* d_ws, size_t ws_size,
                              hipStream_t stream) {
  (void)in_sizes; (void)n_in; (void)out_size; (void)ws_size;
  const float* x   = (const float*)d_in[0];
  const float* wq  = (const float*)d_in[1];
  const float* bq  = (const float*)d_in[2];
  const float* wk  = (const float*)d_in[3];
  const float* bk  = (const float*)d_in[4];
  const float* wv  = (const float*)d_in[5];
  const float* bv  = (const float*)d_in[6];
  const float* d1w = (const float*)d_in[7];
  const float* d1b = (const float*)d_in[8];
  // bn params: indices 9..24 = {bn1,lbn1,lbn2,lbn3} x {g,b,m,v}
  const float* l1w = (const float*)d_in[25];
  const float* l1b = (const float*)d_in[26];
  const float* l2w = (const float*)d_in[27];
  const float* l2b = (const float*)d_in[28];
  const float* l3w = (const float*)d_in[29];
  const float* l3b = (const float*)d_in[30];
  const float* r1w = (const float*)d_in[31];
  const float* r1b = (const float*)d_in[32];
  const float* r2w = (const float*)d_in[37];
  const float* r2b = (const float*)d_in[38];

  char* ws = (char*)d_ws;
  size_t off = 0;
  auto take = [&](size_t bytes) -> char* {
    char* p = ws + off;
    off += (bytes + 255) & ~(size_t)255;
    return p;
  };
  _Float16* wqkv_h  = (_Float16*)take(512 * 256 * 2);
  float*    biasqkv = (float*)take(512 * 4);
  _Float16* d1h     = (_Float16*)take(256 * 256 * 2);
  _Float16* l1h     = (_Float16*)take(256 * 256 * 2);
  _Float16* l3h     = (_Float16*)take(256 * 256 * 2);
  _Float16* l2tap   = (_Float16*)take(9 * 256 * 256 * 2);
  float*    bns     = (float*)take(5 * 256 * 4);
  float*    bnt     = (float*)take(5 * 256 * 4);
  _Float16* xT      = (_Float16*)take((size_t)16384 * 256 * 2);
  _Float16* qb      = (_Float16*)take((size_t)16384 * 128 * 2);
  _Float16* kbf     = (_Float16*)take((size_t)16384 * 128 * 2);
  _Float16* vbf     = (_Float16*)take((size_t)4 * 256 * 4096 * 2);
  float*    featF   = (float*)take((size_t)4 * 256 * 4096 * 4);
  _Float16* featH   = (_Float16*)take((size_t)16384 * 256 * 2);
  _Float16* yNC     = (_Float16*)take((size_t)16384 * 256 * 2);
  _Float16* yl1     = (_Float16*)take((size_t)16384 * 256 * 2);
  _Float16* yl2     = (_Float16*)take((size_t)16384 * 256 * 2);
  float*    yl3F    = (float*)take((size_t)4 * 256 * 4096 * 4);
  float*    pooled  = (float*)take(4 * 256 * 4);
  float*    yr      = (float*)take(4 * 256 * 4);

  PrepArgs pa;
  pa.wq = wq; pa.wk = wk; pa.wv = wv; pa.bq = bq; pa.bk = bk; pa.bv = bv;
  pa.d1w = d1w; pa.l1w = l1w; pa.l2w = l2w; pa.l3w = l3w;
  for (int k = 0; k < 4; ++k) {
    pa.bng[k] = (const float*)d_in[9 + 4 * k + 0];
    pa.bnb[k] = (const float*)d_in[9 + 4 * k + 1];
    pa.bnm[k] = (const float*)d_in[9 + 4 * k + 2];
    pa.bnv[k] = (const float*)d_in[9 + 4 * k + 3];
  }
  pa.rbng = (const float*)d_in[33]; pa.rbnb = (const float*)d_in[34];
  pa.rbnm = (const float*)d_in[35]; pa.rbnv = (const float*)d_in[36];
  pa.wqkv = wqkv_h; pa.d1h = d1h; pa.l1h = l1h; pa.l2tap = l2tap; pa.l3h = l3h;
  pa.biasqkv = biasqkv; pa.bns = bns; pa.bnt = bnt;

  k_prep<<<256, 256, 0, stream>>>(pa);
  k_xpose<<<4096, 256, 0, stream>>>(x, xT);

  // QKV projection: [512 x 256] x [256 x 16384]
  k_gemm<<<dim3(1024, 32), 32, 0, stream>>>(wqkv_h, xT, biasqkv, nullptr, nullptr,
                                            nullptr, nullptr, nullptr,
                                            qb, kbf, vbf, 256, M_QKV);
  // Flash attention -> feat
  k_attn<<<dim3(64, 4), 128, 0, stream>>>(qb, kbf, vbf, featF, featH);
  // dual1 gate: y = feat * sigmoid(relu(bn1(d1(feat))))
  k_gemm<<<dim3(1024, 16), 32, 0, stream>>>(d1h, featH, d1b, bns + 0, bnt + 0,
                                            yNC, nullptr, featH,
                                            nullptr, nullptr, nullptr, 256, M_GATE);
  // SE branch
  k_pool<<<4, 256, 0, stream>>>(yNC, pooled);
  k_se<<<4, 256, 0, stream>>>(pooled, r1w, r1b, bns, bnt, r2w, r2b, yr);
  // local branch: 1x1 -> 3x3 -> 1x1
  k_gemm<<<dim3(1024, 16), 32, 0, stream>>>(l1h, yNC, l1b, bns + 256, bnt + 256,
                                            yl1, nullptr, nullptr,
                                            nullptr, nullptr, nullptr, 256, M_BNRELU_H);
  k_conv3<<<dim3(1024, 16), 32, 0, stream>>>(l2tap, yl1, l2b, bns + 512, bnt + 512, yl2);
  k_gemm<<<dim3(1024, 16), 32, 0, stream>>>(l3h, yl2, l3b, bns + 768, bnt + 768,
                                            nullptr, yl3F, nullptr,
                                            nullptr, nullptr, nullptr, 256, M_BNRELU_F);
  // out = feat * sigmoid(yl + yr)
  k_final<<<4096, 256, 0, stream>>>(featF, yl3F, yr, (float*)d_out);
}
